// AttentionWithContext_89026082111896
// MI455X (gfx1250) — matrix operand order repaired and checked by reading, not compiled, not run on hardware
//
#include <hip/hip_runtime.h>

typedef __attribute__((ext_vector_type(16))) __bf16          v16bf;
typedef __attribute__((ext_vector_type(8)))  float           v8f;

#define B_  64
#define S_  8192
#define F_  128
#define ROWS_PER_WAVE 64
#define WPB (S_ / ROWS_PER_WAVE)   /* waves per batch = 128 */
#define NFRAG (4 * 8 * 32)         /* (kt,nt,lane) fragments */
#define EPS_ 1e-7f

__device__ __forceinline__ unsigned short f2bf(float f) {
  unsigned int u = __builtin_bit_cast(unsigned int, f);
  u += 0x7FFFu + ((u >> 16) & 1u);            // round-to-nearest-even
  return (unsigned short)(u >> 16);
}
__device__ __forceinline__ float bf2f(unsigned short h) {
  return __builtin_bit_cast(float, ((unsigned int)h) << 16);
}

__device__ __forceinline__ float tanh_fast(float x) {
#if __has_builtin(__builtin_amdgcn_tanhf)
  return __builtin_amdgcn_tanhf(x);           // gfx1250 v_tanh_f32 (TRANS op)
#elif __has_builtin(__builtin_amdgcn_tanh_f32)
  return __builtin_amdgcn_tanh_f32(x);
#else
  float cx = fminf(fmaxf(x, -15.0f), 15.0f);
  float e2 = __expf(2.0f * cx);
  return (e2 - 1.0f) / (e2 + 1.0f);
#endif
}

// ---------------------------------------------------------------------------
// Kernel 1: pre-swizzle W (fp32 128x128, row-major) into WMMA B-fragment
// layout, split into bf16 hi + lo.  Fragment index = ((kt*8+nt)*32 + lane)*16.
// For lane l: element i = W[32*kt + 16*(l>>4) + i][ (l&15) + 16*nt ]
// ---------------------------------------------------------------------------
__global__ void prep_kernel(const float* __restrict__ W,
                            unsigned short* __restrict__ bhi,
                            unsigned short* __restrict__ blo) {
  int tid = blockIdx.x * blockDim.x + threadIdx.x;
  if (tid >= NFRAG) return;
  int lane = tid & 31;
  int nt   = (tid >> 5) & 7;
  int kt   = tid >> 8;
  int col   = (lane & 15) + 16 * nt;
  int kbase = 32 * kt + ((lane >> 4) << 4);
#pragma unroll
  for (int i = 0; i < 16; ++i) {
    float w = W[(kbase + i) * F_ + col];
    unsigned short h = f2bf(w);
    bhi[tid * 16 + i] = h;
    blo[tid * 16 + i] = f2bf(w - bf2f(h));
  }
}

// ---------------------------------------------------------------------------
// Kernel 2: fused attention pass.  One wave owns 64 rows (s-values) of one
// batch.  W fragments staged once per block into LDS (64KB of the 320KB WGP
// pool).  The kt loop is kept ROLLED so fragment addresses depend on the
// induction variable: LICM cannot hoist the ds_loads (which previously caused
// 512 VGPRs of live range + scratch spills).  __syncthreads per 16-row group
// is an LDS fence with the same effect.  VGPR live set ~160 -> 4 waves/SIMD.
// Per 16-row group: C(16x128,f32) via 4Kt x 8Nt x 3 v_wmma_f32_16x16x32_bf16
// (split-bf16 fp32 emulation), then tanh/./u/exp epilogue and the weighted
// x-sum (re-read of the just-touched tile: L0/WGP$ hit).
// ---------------------------------------------------------------------------
__global__ void __launch_bounds__(128, 4) attn_kernel(
    const float* __restrict__ x, const unsigned char* __restrict__ mask,
    const float* __restrict__ bvec, const float* __restrict__ uvec,
    const unsigned short* __restrict__ bhi, const unsigned short* __restrict__ blo,
    float* __restrict__ pnum, float* __restrict__ psum) {
  __shared__ __attribute__((aligned(32))) unsigned short lbhi[NFRAG * 16]; // 32KB
  __shared__ __attribute__((aligned(32))) unsigned short lblo[NFRAG * 16]; // 32KB

  // cooperative stage of the pre-swizzled W fragments (L2-resident source)
  for (int i = threadIdx.x; i < (NFRAG * 16) / 8; i += blockDim.x) {
    ((uint4*)lbhi)[i] = ((const uint4*)bhi)[i];
    ((uint4*)lblo)[i] = ((const uint4*)blo)[i];
  }
  __syncthreads();

  const int lane = threadIdx.x & 31;
  const int wave = (int)((blockIdx.x * blockDim.x + threadIdx.x) >> 5);
  const int b     = wave / WPB;
  const int srow0 = (wave % WPB) * ROWS_PER_WAVE;

  // per-lane bias / u for the 8 n-tiles (lane's column = (lane&15)+16*nt)
  const int colbase = lane & 15;
  float bl[8], ul[8];
#pragma unroll
  for (int nt = 0; nt < 8; ++nt) {
    bl[nt] = bvec[colbase + 16 * nt];
    ul[nt] = uvec[colbase + 16 * nt];
  }

  const float* xb = x + (size_t)b * S_ * F_;
  float out_acc[4] = {0.f, 0.f, 0.f, 0.f};
  float sum_e = 0.f;   // this lane-half's running sum of e

  for (int it = 0; it < ROWS_PER_WAVE / 16; ++it) {
    __syncthreads();   // LDS fence: no ds_load may be hoisted above this
    const int rbase = srow0 + it * 16;
    const int arow  = rbase + (lane & 15);

    v8f c[8] = {};   // 8 N-tiles of 16x16 f32 accumulators

#pragma unroll 1     /* keep rolled: fragment addrs depend on kt -> no LICM */
    for (int kt = 0; kt < 4; ++kt) {
      // A-fragment fp32 source: lane<16 -> K {0..15}; lane>=16 -> {16..31}
      const float* xp = xb + (size_t)arow * F_ + 32 * kt + ((lane >> 4) << 4);
      float4 q0 = *(const float4*)(xp);
      float4 q1 = *(const float4*)(xp + 4);
      float4 q2 = *(const float4*)(xp + 8);
      float4 q3 = *(const float4*)(xp + 12);
      float xs[16] = {q0.x, q0.y, q0.z, q0.w, q1.x, q1.y, q1.z, q1.w,
                      q2.x, q2.y, q2.z, q2.w, q3.x, q3.y, q3.z, q3.w};
      v16bf ahi, alo;
#pragma unroll
      for (int i = 0; i < 16; ++i) {          // native v_cvt_pk_bf16_f32
        __bf16 h = (__bf16)xs[i];
        ahi[i] = h;
        alo[i] = (__bf16)(xs[i] - (float)h);
      }

#pragma unroll
      for (int nt = 0; nt < 8; ++nt) {
        int fidx = ((kt * 8 + nt) * 32 + lane) * 16;
        v16bf wh = *(const v16bf*)(lbhi + fidx);   // ds_load_b128 x2
        v16bf wl = *(const v16bf*)(lblo + fidx);
        // fp32-accurate split product: Ah*Wh + Al*Wh + Ah*Wl
        c[nt] = __builtin_amdgcn_wmma_f32_16x16x32_bf16(false, ahi, false, wh,
                                                        (short)0, c[nt], false, false);
        c[nt] = __builtin_amdgcn_wmma_f32_16x16x32_bf16(false, alo, false, wh,
                                                        (short)0, c[nt], false, false);
        c[nt] = __builtin_amdgcn_wmma_f32_16x16x32_bf16(false, ahi, false, wl,
                                                        (short)0, c[nt], false, false);
      }
    }

    // epilogue: per-lane column contribution to ait of row (j + 8*(lane>>4))
    float s[8];
#pragma unroll
    for (int j = 0; j < 8; ++j) {
      float acc = 0.f;
#pragma unroll
      for (int nt = 0; nt < 8; ++nt)
        acc += tanh_fast(c[nt][j] + bl[nt]) * ul[nt];
      s[j] = acc;
    }
    // reduce over the 16 columns held by each lane-half (xor 1,2,4,8 stays in-half)
#pragma unroll
    for (int j = 0; j < 8; ++j) {
#pragma unroll
      for (int m = 1; m <= 8; m <<= 1)
        s[j] += __shfl_xor(s[j], m, 32);
    }

    // e = exp(ait) * mask   (rows j+8*(lane>>4))
    const unsigned char* mp = mask + (size_t)b * S_ + rbase + ((lane >> 4) << 3);
    float e[8];
#pragma unroll
    for (int j = 0; j < 8; ++j) {
      float mv = mp[j] ? 1.0f : 0.0f;
      e[j] = __expf(s[j]) * mv;
      sum_e += e[j];
    }

    // weighted sum: lane covers f = 4*lane..4*lane+3 ; rows 0..7 live in lane 0,
    // rows 8..15 in lane 16.  x tile re-read hits L0/WGP$.
#pragma unroll
    for (int r = 0; r < 16; ++r) {
      float eb = __shfl(e[r & 7], (r < 8) ? 0 : 16, 32);
      const float4 xv = *(const float4*)(xb + (size_t)(rbase + r) * F_ + 4 * lane);
      out_acc[0] += eb * xv.x;
      out_acc[1] += eb * xv.y;
      out_acc[2] += eb * xv.z;
      out_acc[3] += eb * xv.w;
    }
  }

  // deterministic partials: pnum[wave][128], psum[wave]
  float4 ov = make_float4(out_acc[0], out_acc[1], out_acc[2], out_acc[3]);
  *(float4*)(pnum + (size_t)wave * F_ + 4 * lane) = ov;
  float s16 = __shfl(sum_e, 16, 32);
  if (lane == 0) psum[wave] = sum_e + s16;   // lane0 holds half0, lane16 half1
}

// ---------------------------------------------------------------------------
// Kernel 3: fixed-order reduction over the 128 waves of each batch + normalize
// ---------------------------------------------------------------------------
__global__ void finalize_kernel(const float* __restrict__ pnum,
                                const float* __restrict__ psum,
                                float* __restrict__ out) {
  int i = blockIdx.x * blockDim.x + threadIdx.x;
  if (i >= B_ * F_) return;
  int b = i >> 7;
  int f = i & (F_ - 1);
  float acc = 0.f, se = 0.f;
  for (int w = 0; w < WPB; ++w) {
    acc += pnum[((size_t)b * WPB + w) * F_ + f];
    se  += psum[b * WPB + w];
  }
  out[i] = acc / (se + EPS_);
}

extern "C" void kernel_launch(void* const* d_in, const int* in_sizes, int n_in,
                              void* d_out, int out_size, void* d_ws, size_t ws_size,
                              hipStream_t stream) {
  (void)in_sizes; (void)n_in; (void)out_size; (void)ws_size;
  const float*         x    = (const float*)d_in[0];
  const unsigned char* mask = (const unsigned char*)d_in[1];   // jnp bool = 1 byte
  const float*         W    = (const float*)d_in[2];
  const float*         bv   = (const float*)d_in[3];
  const float*         uv   = (const float*)d_in[4];

  // workspace layout
  float* pnum = (float*)d_ws;                               // [B*WPB][F] = 4 MB
  float* psum = pnum + (size_t)B_ * WPB * F_;               // [B*WPB]   = 32 KB
  unsigned short* bhi = (unsigned short*)(psum + B_ * WPB); // 32 KB
  unsigned short* blo = bhi + NFRAG * 16;                   // 32 KB

  prep_kernel<<<4, 256, 0, stream>>>(W, bhi, blo);

  int waves = (B_ * S_) / ROWS_PER_WAVE;                    // 8192 waves
  attn_kernel<<<waves / 4, 128, 0, stream>>>(x, mask, bv, uv, bhi, blo, pnum, psum);

  finalize_kernel<<<(B_ * F_ + 255) / 256, 256, 0, stream>>>(pnum, psum, (float*)d_out);
}
